// SpatioTemporalAttention_37314675868015
// MI455X (gfx1250) — matrix operand order repaired and checked by reading, not compiled, not run on hardware
//
#include <hip/hip_runtime.h>
#include <hip/hip_bf16.h>

typedef __attribute__((ext_vector_type(16))) __bf16 bf16x16;
typedef __attribute__((ext_vector_type(8)))  float  f32x8;
typedef int v4i_vec __attribute__((vector_size(16)));   // matches builtin's V4i

struct BFrag { unsigned u[8]; };

#if defined(__AMDGCN__) && __has_builtin(__builtin_amdgcn_global_load_async_to_lds_b128) && __has_builtin(__builtin_amdgcn_s_wait_asynccnt)
#define USE_ASYNC_LDS 1
#else
#define USE_ASYNC_LDS 0
#endif

__device__ __forceinline__ unsigned short f2bf(float f) {
    unsigned u = __builtin_bit_cast(unsigned, f);
    unsigned r = u + 0x7FFFu + ((u >> 16) & 1u);   // RNE
    return (unsigned short)(r >> 16);
}
__device__ __forceinline__ float bf2f(unsigned short h) {
    unsigned u = ((unsigned)h) << 16;
    return __builtin_bit_cast(float, u);
}
__device__ __forceinline__ unsigned pack2(float a, float b) {
    return (unsigned)f2bf(a) | ((unsigned)f2bf(b) << 16);
}
__device__ __forceinline__ f32x8 wmma_bf16(const BFrag& a, const BFrag& b, f32x8 c) {
    return __builtin_amdgcn_wmma_f32_16x16x32_bf16(
        false, __builtin_bit_cast(bf16x16, a),
        false, __builtin_bit_cast(bf16x16, b),
        (short)0, c, false, false);
}

#if USE_ASYNC_LDS
// Direct global->LDS copy, no VGPR staging (GLOBAL_LOAD_ASYNC_TO_LDS_B128).
// Builtin expects (v4i AS1*, v4i AS3*, imm offset, imm cpol).
// AS3 offset = low 32 bits of the generic LDS address (aperture layout).
typedef __attribute__((address_space(1))) v4i_vec* g_v4i_ptr;
typedef __attribute__((address_space(3))) v4i_vec* l_v4i_ptr;
__device__ __forceinline__ void async_b128(const unsigned* g, unsigned* l) {
    __builtin_amdgcn_global_load_async_to_lds_b128(
        (g_v4i_ptr)(unsigned long long)g,
        (l_v4i_ptr)(unsigned long long)l,
        0, 0);
}
#endif

// ---------------------------------------------------------------------------
// Pack f32 weights [Ksrc][N] -> bf16 k-pair u32 layout [Kdst/2][ldB]
// dst[kp*ldB + n] = (bf16(W[2kp][n]), bf16(W[2kp+1][n]))   (zero padded)
// ---------------------------------------------------------------------------
__global__ __launch_bounds__(256)
void pack_w_kernel(const float* __restrict__ src, unsigned* __restrict__ dst,
                   int Ksrc, int N, int ldB, size_t total) {
    size_t idx = (size_t)blockIdx.x * 256 + threadIdx.x;
    if (idx >= total) return;
    int kp = (int)(idx / ldB);
    int n  = (int)(idx % ldB);
    int k0 = 2 * kp, k1 = 2 * kp + 1;
    float a = (n < N && k0 < Ksrc) ? src[(size_t)k0 * N + n] : 0.f;
    float b = (n < N && k1 < Ksrc) ? src[(size_t)k1 * N + n] : 0.f;
    dst[idx] = pack2(a, b);
}

// ---------------------------------------------------------------------------
// CPB relative-position bias MLP: bias[h][i][j], h<8, i,j<16
// ---------------------------------------------------------------------------
__global__ __launch_bounds__(256)
void cpb_bias_kernel(const float* __restrict__ w1, const float* __restrict__ b1,
                     const float* __restrict__ w2, const float* __restrict__ b2,
                     const float* __restrict__ w3, const float* __restrict__ b3,
                     float* __restrict__ bias) {
    __shared__ float h1[256];
    __shared__ float h2[256];
    int ij = blockIdx.x;
    int i = ij >> 4, j = ij & 15;
    float rel = (float)(i - j);
    float sg = (rel > 0.f) ? 1.f : ((rel < 0.f) ? -1.f : 0.f);
    float val = sg * logf(fabsf(rel) + 1.f);
    int t = threadIdx.x;
    float a = val * w1[t] + b1[t];
    h1[t] = a / (1.f + expf(-a));                 // silu
    __syncthreads();
    float acc = b2[t];
    for (int k = 0; k < 256; ++k) acc += h1[k] * w2[k * 256 + t];
    h2[t] = acc / (1.f + expf(-acc));
    __syncthreads();
    if (t < 8) {
        float o = b3[t];
        for (int k = 0; k < 256; ++k) o += h2[k] * w3[k * 8 + t];
        bias[(t * 16 + i) * 16 + j] = o;
    }
}

// ---------------------------------------------------------------------------
// x (b,c,t,h,w) f32 -> xt rows [(b,hw,t)][c] f32 (no embeddings: residual src)
// ---------------------------------------------------------------------------
__global__ __launch_bounds__(256)
void transpose_in_kernel(const float* __restrict__ x, float* __restrict__ xt) {
    __shared__ float tile[32][33];
    int bt = blockIdx.x; int b = bt >> 4; int t = bt & 15;
    int c0 = blockIdx.y << 5; int hw0 = blockIdx.z << 5;
    int xi = threadIdx.x & 31; int yi = threadIdx.x >> 5;
#pragma unroll
    for (int ii = 0; ii < 4; ++ii) {
        int c = c0 + yi + (ii << 3);
        tile[yi + (ii << 3)][xi] =
            x[((((size_t)b * 512 + c) * 16 + t) << 10) + hw0 + xi];
    }
    __syncthreads();
#pragma unroll
    for (int ii = 0; ii < 4; ++ii) {
        int hw = hw0 + yi + (ii << 3);
        xt[(((size_t)(b << 10) + hw) * 16 + t) * 512 + c0 + xi] =
            tile[xi][yi + (ii << 3)];
    }
}

// ---------------------------------------------------------------------------
// Per-pixel time LayerNorm of (xt + pos_emb + fr_emb) over t=16 -> xn bf16
// ---------------------------------------------------------------------------
__global__ __launch_bounds__(256)
void time_norm_kernel(const float* __restrict__ xt, const float* __restrict__ pos,
                      const float* __restrict__ fre, const int* __restrict__ frp,
                      const float* __restrict__ g, unsigned short* __restrict__ xn) {
    int pix = blockIdx.x;            // 2048
    int fr = frp[0] - 1;
    for (int c0 = 0; c0 < 512; c0 += 256) {
        int c = c0 + threadIdx.x;
        float v[16]; float s = 0.f;
#pragma unroll
        for (int t = 0; t < 16; ++t) {
            float e = xt[((size_t)pix * 16 + t) * 512 + c]
                    + pos[t * 512 + c] + fre[fr * 512 + c];
            v[t] = e; s += e;
        }
        float mean = s * (1.f / 16.f);
        float q = 0.f;
#pragma unroll
        for (int t = 0; t < 16; ++t) { float d = v[t] - mean; q += d * d; }
        float rs = rsqrtf(fmaxf(q * (1.f / 16.f), 1e-5f)) * g[c];
#pragma unroll
        for (int t = 0; t < 16; ++t)
            xn[((size_t)pix * 16 + t) * 512 + c] = f2bf((v[t] - mean) * rs);
    }
}

// ---------------------------------------------------------------------------
// Generic bf16 WMMA GEMM:  C[M x N] = A[M x 2*Kp] (bf16) @ B, fused epilogue.
// A: u32 k-pair rows, stride Kp.  B: packed weights [Kp][ldB] u32.
// BM=BN=128, BK=32, 128 threads (4 waves), wave tile 64x64 (16 wmma/k-step).
// Double-buffered async (GLOBAL_LOAD_ASYNC_TO_LDS_B128 + s_wait_asynccnt).
// NOTE: B tile over-read past N is allowed (epilogue masks gn<N); the packed
// weight buffers are allocated with tail padding so reads stay in-bounds.
// ---------------------------------------------------------------------------
__global__ __launch_bounds__(128)
void gemm_bf16_kernel(const unsigned* __restrict__ A,
                      const unsigned* __restrict__ Bw,
                      int N, int Kp, int ldB,
                      const float* __restrict__ resid, int ldR,
                      float* __restrict__ outF,
                      unsigned short* __restrict__ outH, int ldO) {
    __shared__ unsigned lsA[2][128][20];   // 16 used; stride 20 -> conflict-free
    __shared__ unsigned lsB[2][16][132];   // 128 used; stride 132 -> conflict-free

    const int tid = threadIdx.x;
    const int n0 = blockIdx.x << 7;
    const size_t m0 = (size_t)blockIdx.y << 7;
    const int wave = tid >> 5, lane = tid & 31;
    const int wm = wave >> 1, wn = wave & 1;
    const int half = lane >> 4, lm = lane & 15;
    const int bi = tid >> 3;
    const int jc = (tid & 7) << 4;
    const int nK = Kp >> 4;

    f32x8 acc[4][4];
#pragma unroll
    for (int i = 0; i < 4; ++i)
#pragma unroll
        for (int j = 0; j < 4; ++j) acc[i][j] = {};

    auto load_tile = [&](int ks, int buf) {
        const unsigned* Arow = A + (m0 + tid) * (size_t)Kp + (ks << 4);
        const unsigned* Brow = Bw + (size_t)((ks << 4) + bi) * ldB + n0 + jc;
#if USE_ASYNC_LDS
#pragma unroll
        for (int e = 0; e < 4; ++e)
            async_b128(Arow + (e << 2), &lsA[buf][tid][e << 2]);
#pragma unroll
        for (int e = 0; e < 4; ++e)
            async_b128(Brow + (e << 2), &lsB[buf][bi][jc + (e << 2)]);
#else
        uint4 a0 = ((const uint4*)Arow)[0];
        uint4 a1 = ((const uint4*)Arow)[1];
        uint4 a2 = ((const uint4*)Arow)[2];
        uint4 a3 = ((const uint4*)Arow)[3];
        uint4 b0 = ((const uint4*)Brow)[0];
        uint4 b1 = ((const uint4*)Brow)[1];
        uint4 b2 = ((const uint4*)Brow)[2];
        uint4 b3 = ((const uint4*)Brow)[3];
        *(uint4*)&lsA[buf][tid][0]  = a0;
        *(uint4*)&lsA[buf][tid][4]  = a1;
        *(uint4*)&lsA[buf][tid][8]  = a2;
        *(uint4*)&lsA[buf][tid][12] = a3;
        *(uint4*)&lsB[buf][bi][jc]      = b0;
        *(uint4*)&lsB[buf][bi][jc + 4]  = b1;
        *(uint4*)&lsB[buf][bi][jc + 8]  = b2;
        *(uint4*)&lsB[buf][bi][jc + 12] = b3;
#endif
    };

    load_tile(0, 0);

    for (int ks = 0; ks < nK; ++ks) {
        const int cur = ks & 1;
        if (ks + 1 < nK) {
            load_tile(ks + 1, cur ^ 1);
#if USE_ASYNC_LDS
            __builtin_amdgcn_s_wait_asynccnt(8);   // tile ks landed, ks+1 in flight
#endif
        } else {
#if USE_ASYNC_LDS
            __builtin_amdgcn_s_wait_asynccnt(0);
#endif
        }
        __syncthreads();

        BFrag af[4], bfr[4];
#pragma unroll
        for (int mt = 0; mt < 4; ++mt) {
            int m = (wm << 6) + (mt << 4) + lm;
#pragma unroll
            for (int g = 0; g < 8; ++g) {
                int kp = ((g & 4) << 1) + (half << 2) + (g & 3);
                af[mt].u[g] = lsA[cur][m][kp];
            }
        }
#pragma unroll
        for (int nt = 0; nt < 4; ++nt) {
            int n = (wn << 6) + (nt << 4) + lm;
#pragma unroll
            for (int g = 0; g < 8; ++g)
                bfr[nt].u[g] = lsB[cur][((g & 4) << 1) + (half << 2) + (g & 3)][n];
        }
#pragma unroll
        for (int mt = 0; mt < 4; ++mt)
#pragma unroll
            for (int nt = 0; nt < 4; ++nt)
                acc[mt][nt] = wmma_bf16(af[mt], bfr[nt], acc[mt][nt]);
        __syncthreads();
    }

    // epilogue: optional residual, f32 and/or bf16 stores
#pragma unroll
    for (int mt = 0; mt < 4; ++mt) {
#pragma unroll
        for (int nt = 0; nt < 4; ++nt) {
            int gn = n0 + (wn << 6) + (nt << 4) + lm;
            if (gn < N) {
#pragma unroll
                for (int r = 0; r < 8; ++r) {
                    size_t gm = m0 + (wm << 6) + (mt << 4) + r + (half << 3);
                    float v = acc[mt][nt][r];
                    if (resid) v += resid[gm * (size_t)ldR + gn];
                    if (outF) outF[gm * (size_t)ldO + gn] = v;
                    if (outH) outH[gm * (size_t)ldO + gn] = f2bf(v);
                }
            }
        }
    }
}

// ---------------------------------------------------------------------------
// Attention: one wave per (pixel, head). q(16x64)·k^T via 2 WMMA (K=64),
// softmax f32 (shfl within 16-lane halves), attn (padded K=32) · v via WMMA.
// ---------------------------------------------------------------------------
__global__ __launch_bounds__(128)
void attn_kernel(const unsigned* __restrict__ Q,        // qb u32 [32768][256]
                 const unsigned* __restrict__ KV,       // kvb u32 [32768][512]
                 const unsigned short* __restrict__ KVh,// kvb u16 view
                 const float* __restrict__ bias,        // [8][16][16]
                 unsigned short* __restrict__ O) {      // [32768][512]
    __shared__ unsigned short lat[4][16][18];
    const int wave = threadIdx.x >> 5, lane = threadIdx.x & 31;
    const int id = blockIdx.x * 4 + wave;
    const int pix = id >> 3, head = id & 7;
    const int half = lane >> 4, lm = lane & 15;
    const float scale = 0.125f;   // 64^-0.5

    BFrag qf[2], kf[2];
#pragma unroll
    for (int ch = 0; ch < 2; ++ch) {
        size_t qrow = ((size_t)pix * 16 + lm) * 256 + head * 32 + ch * 16;
        size_t krow = ((size_t)pix * 16 + lm) * 512 + head * 32 + ch * 16;
#pragma unroll
        for (int g = 0; g < 8; ++g) {
            int kpa = ((g & 4) << 1) + (half << 2) + (g & 3);  // A layout
            qf[ch].u[g] = Q[qrow + kpa];
            int kpb = ((g & 4) << 1) + (half << 2) + (g & 3);  // B layout
            kf[ch].u[g] = KV[krow + kpb];
        }
    }
    f32x8 c = {};
    c = wmma_bf16(qf[0], kf[0], c);
    c = wmma_bf16(qf[1], kf[1], c);

    // scale + bias + softmax over j (16 lanes of this half)
    float p[8];
#pragma unroll
    for (int r = 0; r < 8; ++r) {
        int i = r + (half << 3);
        float v = c[r] * scale + bias[((head << 4) + i) * 16 + lm];
        float mx = v;
#pragma unroll
        for (int m = 1; m < 16; m <<= 1) mx = fmaxf(mx, __shfl_xor(mx, m, 32));
        float e = expf(v - mx);
        float s = e;
#pragma unroll
        for (int m = 1; m < 16; m <<= 1) s += __shfl_xor(s, m, 32);
        p[r] = e / s;
    }
    // stage attn in LDS (C layout -> A layout cross-lane transpose)
#pragma unroll
    for (int r = 0; r < 8; ++r)
        lat[wave][r + (half << 3)][lm] = f2bf(p[r]);
    __syncthreads();

    BFrag af;
#pragma unroll
    for (int g = 0; g < 4; ++g) {                      // j = 0..15 valid
        int j0 = (half << 3) + (g << 1);
        af.u[g] = *(const unsigned*)&lat[wave][lm][j0];
    }
#pragma unroll
    for (int g = 4; g < 8; ++g) af.u[g] = 0u;          // j >= 16 padded zero

#pragma unroll
    for (int nt = 0; nt < 4; ++nt) {
        BFrag vf;
#pragma unroll
        for (int g = 0; g < 8; ++g) {
            if (g < 4) {
                int j0 = (half << 3) + (g << 1);
                size_t base = ((size_t)pix * 16 + j0) * 1024 + 512
                            + head * 64 + nt * 16 + lm;
                unsigned lo = KVh[base];
                unsigned hi = KVh[base + 1024];
                vf.u[g] = lo | (hi << 16);
            } else vf.u[g] = 0u;
        }
        f32x8 o = {};
        o = wmma_bf16(af, vf, o);
#pragma unroll
        for (int r = 0; r < 8; ++r)
            O[((size_t)pix * 16 + r + (half << 3)) * 512
              + head * 64 + nt * 16 + lm] = f2bf(o[r]);
    }
}

// ---------------------------------------------------------------------------
// SwiGLU: y1 * gelu_exact(gate)
// ---------------------------------------------------------------------------
__global__ __launch_bounds__(256)
void swiglu_kernel(const unsigned short* __restrict__ y,
                   unsigned short* __restrict__ o, size_t total) {
    size_t idx = (size_t)blockIdx.x * 256 + threadIdx.x;
    if (idx >= total) return;
    size_t r = idx / 1365; int cidx = (int)(idx % 1365);
    float a = bf2f(y[r * 2730 + cidx]);
    float g = bf2f(y[r * 2730 + 1365 + cidx]);
    float ge = 0.5f * g * (1.f + erff(g * 0.70710678118654752f));
    o[r * 1365 + cidx] = f2bf(a * ge);
}

// ---------------------------------------------------------------------------
// Temporal shift of channels >= 683 + channel LayerNorm(1365) -> yn bf16[1376]
// ---------------------------------------------------------------------------
__global__ __launch_bounds__(256)
void shiftnorm_kernel(const unsigned short* __restrict__ ysw,
                      const float* __restrict__ cg,
                      unsigned short* __restrict__ yn) {
    __shared__ float rs_[256];
    __shared__ float rq_[256];
    int r = blockIdx.x; int t = r & 15; int tid = threadIdx.x;
    float v[6]; float s = 0.f, sq = 0.f;
#pragma unroll
    for (int e = 0; e < 6; ++e) {
        int c = tid + (e << 8);
        float x = 0.f;
        if (c < 1365) {
            if (c < 683) x = bf2f(ysw[(size_t)r * 1365 + c]);
            else if (t > 0) x = bf2f(ysw[(size_t)(r - 1) * 1365 + c]);
        }
        v[e] = x; s += x; sq += x * x;
    }
    rs_[tid] = s; rq_[tid] = sq; __syncthreads();
    for (int w = 128; w > 0; w >>= 1) {
        if (tid < w) { rs_[tid] += rs_[tid + w]; rq_[tid] += rq_[tid + w]; }
        __syncthreads();
    }
    float mean = rs_[0] * (1.f / 1365.f);
    float var  = rq_[0] * (1.f / 1365.f) - mean * mean;
    float rstd = rsqrtf(fmaxf(var, 1e-5f));
#pragma unroll
    for (int e = 0; e < 6; ++e) {
        int c = tid + (e << 8);
        if (c < 1365)
            yn[(size_t)r * 1376 + c] = f2bf((v[e] - mean) * rstd * cg[c]);
    }
    if (tid < 11) yn[(size_t)r * 1376 + 1365 + tid] = 0;   // K padding
}

// ---------------------------------------------------------------------------
// rows [(b,hw,t)][c] f32 -> out (b,c,t,h,w)
// ---------------------------------------------------------------------------
__global__ __launch_bounds__(256)
void transpose_out_kernel(const float* __restrict__ rows, float* __restrict__ out) {
    __shared__ float tile[32][33];
    int bt = blockIdx.x; int b = bt >> 4; int t = bt & 15;
    int c0 = blockIdx.y << 5; int hw0 = blockIdx.z << 5;
    int xi = threadIdx.x & 31; int yi = threadIdx.x >> 5;
#pragma unroll
    for (int ii = 0; ii < 4; ++ii) {
        int hw = hw0 + yi + (ii << 3);
        tile[xi][yi + (ii << 3)] =
            rows[(((size_t)(b << 10) + hw) * 16 + t) * 512 + c0 + xi];
    }
    __syncthreads();
#pragma unroll
    for (int ii = 0; ii < 4; ++ii) {
        int c = c0 + yi + (ii << 3);
        out[((((size_t)b * 512 + c) * 16 + t) << 10) + hw0 + xi] =
            tile[yi + (ii << 3)][xi];
    }
}

// ---------------------------------------------------------------------------
extern "C" void kernel_launch(void* const* d_in, const int* in_sizes, int n_in,
                              void* d_out, int out_size, void* d_ws, size_t ws_size,
                              hipStream_t stream) {
    const float* x       = (const float*)d_in[0];
    const int*   frate   = (const int*)  d_in[1];
    const float* g_norm  = (const float*)d_in[2];
    const float* Wq      = (const float*)d_in[3];
    const float* Wkv     = (const float*)d_in[4];
    const float* Wo      = (const float*)d_in[5];
    const float* pos_emb = (const float*)d_in[6];
    const float* fr_emb  = (const float*)d_in[7];
    const float* cpb_w1  = (const float*)d_in[8];
    const float* cpb_b1  = (const float*)d_in[9];
    const float* cpb_w2  = (const float*)d_in[10];
    const float* cpb_b2  = (const float*)d_in[11];
    const float* cpb_w3  = (const float*)d_in[12];
    const float* cpb_b3  = (const float*)d_in[13];
    const float* Win     = (const float*)d_in[14];
    const float* chan_g  = (const float*)d_in[15];
    const float* Wout    = (const float*)d_in[16];
    float* out = (float*)d_out;

    const size_t M = 32768;   // b*h*w*t rows
    char* ws = (char*)d_ws;
    size_t off = 0;
    auto take = [&](size_t bytes) -> char* {
        char* p = ws + off; off = (off + bytes + 255) & ~(size_t)255; return p;
    };
    float*          xt    = (float*)         take(M * 512 * 4);
    unsigned short* xn    = (unsigned short*)take(M * 512 * 2);
    unsigned short* qb    = (unsigned short*)take(M * 512 * 2);
    unsigned short* kvb   = (unsigned short*)take(M * 1024 * 2);
    unsigned short* ob    = (unsigned short*)take(M * 512 * 2);
    float*          xt2   = (float*)         take(M * 512 * 4);
    unsigned short* xt2b  = (unsigned short*)take(M * 512 * 2);
    unsigned*       wqp   = (unsigned*)      take((size_t)256 * 512 * 4);
    unsigned*       wkvp  = (unsigned*)      take((size_t)256 * 1024 * 4);
    unsigned*       wop   = (unsigned*)      take((size_t)256 * 512 * 4);
    unsigned*       winp  = (unsigned*)      take((size_t)256 * 2732 * 4 + 1024); // +tail pad (over-read)
    unsigned*       woutp = (unsigned*)      take((size_t)688 * 512 * 4);
    float*          biasb = (float*)         take(8 * 16 * 16 * 4);
    unsigned short* yraw  = (unsigned short*)take(M * 2730 * 2);
    unsigned short* ysw   = (unsigned short*)take(M * 1365 * 2);
    unsigned short* yn    = (unsigned short*)take(M * 1376 * 2);
    float*          rows  = (float*)         take(M * 512 * 4);

    auto packW = [&](const float* src, unsigned* dst, int Ksrc, int Kdst,
                     int N, int ldB) {
        size_t total = (size_t)(Kdst / 2) * ldB;
        pack_w_kernel<<<(unsigned)((total + 255) / 256), 256, 0, stream>>>(
            src, dst, Ksrc, N, ldB, total);
    };
    packW(Wq,   wqp,   512,  512,  512,  512);
    packW(Wkv,  wkvp,  512,  512,  1024, 1024);
    packW(Wo,   wop,   512,  512,  512,  512);
    packW(Win,  winp,  512,  512,  2730, 2732);
    packW(Wout, woutp, 1365, 1376, 512,  512);

    cpb_bias_kernel<<<256, 256, 0, stream>>>(cpb_w1, cpb_b1, cpb_w2, cpb_b2,
                                             cpb_w3, cpb_b3, biasb);

    transpose_in_kernel<<<dim3(32, 16, 32), 256, 0, stream>>>(x, xt);
    time_norm_kernel<<<2048, 256, 0, stream>>>(xt, pos_emb, fr_emb, frate,
                                               g_norm, xn);

    // Q = xn @ Wq
    gemm_bf16_kernel<<<dim3(4, 256), 128, 0, stream>>>(
        (const unsigned*)xn, wqp, 512, 256, 512,
        nullptr, 0, nullptr, qb, 512);
    // KV = xn @ Wkv
    gemm_bf16_kernel<<<dim3(8, 256), 128, 0, stream>>>(
        (const unsigned*)xn, wkvp, 1024, 256, 1024,
        nullptr, 0, nullptr, kvb, 1024);

    attn_kernel<<<4096, 128, 0, stream>>>(
        (const unsigned*)qb, (const unsigned*)kvb, kvb, biasb, ob);

    // xt2 = o @ Wo + xt   (f32 + bf16 copies)
    gemm_bf16_kernel<<<dim3(4, 256), 128, 0, stream>>>(
        (const unsigned*)ob, wop, 512, 256, 512,
        xt, 512, xt2, xt2b, 512);

    // yraw = xt2 @ Win  (N = 2730)
    gemm_bf16_kernel<<<dim3(22, 256), 128, 0, stream>>>(
        (const unsigned*)xt2b, winp, 2730, 256, 2732,
        nullptr, 0, nullptr, yraw, 2730);

    size_t swl = M * 1365;
    swiglu_kernel<<<(unsigned)((swl + 255) / 256), 256, 0, stream>>>(yraw, ysw, swl);
    shiftnorm_kernel<<<32768, 256, 0, stream>>>(ysw, chan_g, yn);

    // rows = yn @ Wout + xt2   (K = 1376 padded)
    gemm_bf16_kernel<<<dim3(4, 256), 128, 0, stream>>>(
        (const unsigned*)yn, woutp, 512, 688, 512,
        xt2, 512, rows, nullptr, 512);

    transpose_out_kernel<<<dim3(32, 16, 32), 256, 0, stream>>>(rows, out);
}
